// DenseRetriever_15006615733207
// MI455X (gfx1250) — matrix operand order repaired and checked by reading, not compiled, not run on hardware
//
#include <hip/hip_runtime.h>

typedef __attribute__((ext_vector_type(16))) __bf16 v16bf;
typedef __attribute__((ext_vector_type(8)))  __bf16 v8bf;
typedef __attribute__((ext_vector_type(4)))  __bf16 v4bf;
typedef __attribute__((ext_vector_type(8)))  float  v8f;
typedef __attribute__((ext_vector_type(4)))  float  fvec4;

#define DOC_TILE 128
#define KCHUNK   128
#define NQ       64
#define DIM      768
#define TOPK_CAP 10

struct Cand { float s; int i; };

// ---------------------------------------------------------------------------
// Kernel 1: per-query inverse L2 norm + bf16 copy of the query matrix.
// ---------------------------------------------------------------------------
__global__ __launch_bounds__(256)
void qprep_kernel(const float* __restrict__ q, __bf16* __restrict__ qn,
                  float* __restrict__ qinv) {
  const int row = blockIdx.x;
  const float* qr = q + (size_t)row * DIM;
  __shared__ float red[256];
  float ss = 0.f;
  for (int k = threadIdx.x; k < DIM; k += 256) { float v = qr[k]; ss += v * v; }
  red[threadIdx.x] = ss;
  __syncthreads();
  for (int s = 128; s > 0; s >>= 1) {
    if (threadIdx.x < s) red[threadIdx.x] += red[threadIdx.x + s];
    __syncthreads();
  }
  if (threadIdx.x == 0) qinv[row] = 1.0f / fmaxf(sqrtf(red[0]), 1e-12f);
  for (int k = threadIdx.x; k < DIM; k += 256)
    qn[(size_t)row * DIM + k] = (__bf16)qr[k];
}

// Register-resident unrolled top-10 insertion (no dynamic array indexing).
__device__ __forceinline__ void topk_insert(float v, int id,
                                            float (&bs)[TOPK_CAP],
                                            int (&bi)[TOPK_CAP]) {
#pragma unroll
  for (int r = 0; r < TOPK_CAP; ++r) {
    bool gt = v > bs[r];
    float ts = bs[r]; int ti = bi[r];
    bs[r] = gt ? v : ts;  bi[r] = gt ? id : ti;
    v = gt ? ts : v;      id = gt ? ti : id;
  }
}

// ---------------------------------------------------------------------------
// Kernel 2: fused bf16 WMMA GEMM + per-doc norms + per-tile top-10.
// One block = 128 docs x 64 queries. 8 waves; each wave owns a 64x16 strip
// as 4 accumulator tiles of v_wmma_f32_16x16x32_bf16.
// ---------------------------------------------------------------------------
__global__ __launch_bounds__(256)
void score_topk_kernel(const float* __restrict__ docs,
                       const __bf16* __restrict__ qn,
                       const float* __restrict__ qinv,
                       Cand* __restrict__ cand,
                       int N, int nblocks) {
  __shared__ __bf16 sD[DOC_TILE][KCHUNK];   // 32 KB doc tile (bf16, K chunk)
  __shared__ __bf16 sQ[NQ][KCHUNK];         // 16 KB query chunk (bf16)
  __shared__ float  sScores[NQ][DOC_TILE];  // 32 KB score tile
  __shared__ float  sNorm[DOC_TILE];
  __shared__ float  sQinv[NQ];

  const int tid  = threadIdx.x;
  const int lane = tid & 31;
  const int wave = tid >> 5;
  const int nlow = lane & 15;
  const int hi   = lane >> 4;           // 0 or 1 (lane group)
  const int docBase = blockIdx.x * DOC_TILE;

  if (tid < DOC_TILE) sNorm[tid] = 0.f;
  if (tid < NQ)       sQinv[tid] = qinv[tid];
  __syncthreads();

  v8f acc[4] = {};

  const unsigned sQbase = (unsigned)(unsigned long long)(&sQ[0][0]);
  const unsigned long long qnbase = (unsigned long long)qn;

  for (int kc = 0; kc < DIM / KCHUNK; ++kc) {
    // --- async copy of the bf16 query chunk: 64 rows x 256 B (1024 x 16 B) ---
    for (int i = tid; i < NQ * (KCHUNK / 8); i += 256) {
      int row = i >> 4, seg = i & 15;
      unsigned long long ga =
          qnbase + (unsigned long long)(row * DIM + kc * KCHUNK) * 2ull +
          (unsigned long long)seg * 16ull;
      unsigned la = sQbase + (unsigned)(row * KCHUNK * 2 + seg * 16);
      asm volatile("global_load_async_to_lds_b128 %0, %1, off"
                   :: "v"(la), "v"(ga) : "memory");
    }

    // --- one prefetch per thread: cover the whole next 64 KB chunk at 256 B ---
    if (kc + 1 < DIM / KCHUNK) {
      int pdoc = docBase + (tid >> 1);          // 2 threads per doc row
      if (pdoc < N)
        __builtin_prefetch(docs + (size_t)pdoc * DIM + (kc + 1) * KCHUNK +
                               (tid & 1) * 64,
                           0, 1);               // global_prefetch_b8
    }

    // --- stream fp32 doc chunk (non-temporal), convert to bf16, norms ---
    for (int i = tid; i < DOC_TILE * (KCHUNK / 4); i += 256) {
      int doc = i >> 5;
      int k4  = (i & 31) * 4;
      int dg  = docBase + doc;
      fvec4 v = {};
      if (dg < N) {
        const fvec4* gp =
            (const fvec4*)(docs + (size_t)dg * DIM + kc * KCHUNK + k4);
        v = __builtin_nontemporal_load(gp);     // th:TH_LOAD_NT — stream once
      }
      v4bf b;
      b[0] = (__bf16)v[0]; b[1] = (__bf16)v[1];
      b[2] = (__bf16)v[2]; b[3] = (__bf16)v[3];
      *(v4bf*)&sD[doc][k4] = b;
      atomicAdd(&sNorm[doc], v[0] * v[0] + v[1] * v[1] + v[2] * v[2] + v[3] * v[3]);
    }

    asm volatile("s_wait_asynccnt 0" ::: "memory");
    __syncthreads();

    // --- WMMA over this K chunk: 4 k-steps; batch all fragment loads, then
    //     issue the 4 WMMAs back-to-back (partial dscnt waits, one clause) ---
#pragma unroll
    for (int kk = 0; kk < KCHUNK; kk += 32) {
      // B fragment: doc = wave*16 + nlow, 16 contiguous K values per lane.
      v16bf bfrag = *(const v16bf*)&sD[wave * 16 + nlow][kk + hi * 16];
      v16bf af[4];
#pragma unroll
      for (int qt = 0; qt < 4; ++qt) {
        // A fragment: row = qt*16 + nlow; 16 contiguous K values per lane (two 8-element chunks).
        v8bf a0 = *(const v8bf*)&sQ[qt * 16 + nlow][kk + hi * 16];
        v8bf a1 = *(const v8bf*)&sQ[qt * 16 + nlow][kk + hi * 16 + 8];
#pragma unroll
        for (int e = 0; e < 8; ++e) { af[qt][e] = a0[e]; af[qt][e + 8] = a1[e]; }
      }
#pragma unroll
      for (int qt = 0; qt < 4; ++qt)
        acc[qt] = __builtin_amdgcn_wmma_f32_16x16x32_bf16(
            false, af[qt], false, bfrag, (short)0, acc[qt], false, false);
    }
    __syncthreads();
  }

  // --- scale by 1/(|q|*|d|) and stage scores in LDS ---
  {
    int doc = wave * 16 + nlow;
    float dinv = 1.0f / fmaxf(sqrtf(sNorm[doc]), 1e-12f);
#pragma unroll
    for (int qt = 0; qt < 4; ++qt) {
#pragma unroll
      for (int r = 0; r < 8; ++r) {
        int qrow = qt * 16 + r + hi * 8;   // C/D layout: M = r + 8*lanegroup
        sScores[qrow][doc] = acc[qt][r] * sQinv[qrow] * dinv;
      }
    }
  }
  __syncthreads();

  // --- per-query top-10 over this 128-doc tile ---
  if (tid < NQ) {
    float bs[TOPK_CAP]; int bi[TOPK_CAP];
#pragma unroll
    for (int r = 0; r < TOPK_CAP; ++r) { bs[r] = -__builtin_inff(); bi[r] = -1; }
    for (int j = 0; j < DOC_TILE; ++j) {
      int dg = docBase + j;
      if (dg >= N) break;
      topk_insert(sScores[tid][j], dg, bs, bi);
    }
    Cand* out = cand + ((size_t)tid * nblocks + blockIdx.x) * TOPK_CAP;
#pragma unroll
    for (int r = 0; r < TOPK_CAP; ++r) { out[r].s = bs[r]; out[r].i = bi[r]; }
  }
}

// ---------------------------------------------------------------------------
// Kernel 3: reduce per-block candidates to the final top-K per query.
// ---------------------------------------------------------------------------
__global__ __launch_bounds__(256)
void topk_final_kernel(const Cand* __restrict__ cand, int nblocks, int K,
                       int* __restrict__ out_idx, float* __restrict__ out_scr) {
  const int q = blockIdx.x;
  const int tid = threadIdx.x;
  const int M = nblocks * TOPK_CAP;
  const Cand* base = cand + (size_t)q * M;

  float bs[TOPK_CAP]; int bi[TOPK_CAP];
#pragma unroll
  for (int r = 0; r < TOPK_CAP; ++r) { bs[r] = -__builtin_inff(); bi[r] = -1; }
  for (int i = tid; i < M; i += 256) {
    Cand c = base[i];
    if (c.i >= 0) topk_insert(c.s, c.i, bs, bi);
  }

  __shared__ float Ls[256][TOPK_CAP];
  __shared__ int   Li[256][TOPK_CAP];
  __shared__ int   head[256];
  __shared__ float rv[256];
  __shared__ int   rt[256];
#pragma unroll
  for (int r = 0; r < TOPK_CAP; ++r) { Ls[tid][r] = bs[r]; Li[tid][r] = bi[r]; }
  head[tid] = 0;
  __syncthreads();

  for (int r = 0; r < K; ++r) {
    int h = head[tid];
    rv[tid] = (h < TOPK_CAP) ? Ls[tid][h] : -__builtin_inff();
    rt[tid] = tid;
    __syncthreads();
    for (int s = 128; s > 0; s >>= 1) {
      if (tid < s && rv[tid + s] > rv[tid]) { rv[tid] = rv[tid + s]; rt[tid] = rt[tid + s]; }
      __syncthreads();
    }
    if (tid == 0) {
      int w = rt[0], hw = head[w];
      out_idx[q * K + r] = Li[w][hw];
      out_scr[q * K + r] = Ls[w][hw];
      head[w] = hw + 1;
    }
    __syncthreads();
  }
}

// ---------------------------------------------------------------------------
extern "C" void kernel_launch(void* const* d_in, const int* in_sizes, int n_in,
                              void* d_out, int out_size, void* d_ws, size_t ws_size,
                              hipStream_t stream) {
  const float* qemb = (const float*)d_in[0];
  const float* docs = (const float*)d_in[1];

  const int Q = in_sizes[0] / DIM;          // 64
  const int N = in_sizes[1] / DIM;          // 500000
  const int K = out_size / (2 * Q);         // 10
  const int nblocks = (N + DOC_TILE - 1) / DOC_TILE;

  char* ws = (char*)d_ws;
  __bf16* qn  = (__bf16*)ws;                               // Q*DIM*2 bytes
  float* qinv = (float*)(ws + (size_t)Q * DIM * 2);        // Q*4 bytes
  Cand* cand  = (Cand*)(ws + (size_t)Q * DIM * 2 + 256);   // Q*nblocks*10*8 bytes

  int*   out_idx = (int*)d_out;
  float* out_scr = (float*)d_out + (size_t)Q * K;

  qprep_kernel<<<Q, 256, 0, stream>>>(qemb, qn, qinv);
  score_topk_kernel<<<nblocks, 256, 0, stream>>>(docs, qn, qinv, cand, N, nblocks);
  topk_final_kernel<<<Q, 256, 0, stream>>>(cand, nblocks, K, out_idx, out_scr);
}
